// RelativeMultiHeadAttention_6871947673612
// MI455X (gfx1250) — matrix operand order repaired and checked by reading, not compiled, not run on hardware
//
#include <hip/hip_runtime.h>
#include <hip/hip_bf16.h>
#include <cstdint>
#include <cstddef>

#define D_MODEL 1024
#define N_HEADS 16
#define HEAD_DIM 64
#define QLEN 1024
#define MLEN 1024
#define RLEN (QLEN + MLEN)   /* klen = 2048 */
#define BSZ 2
#define LN_EPS 1e-5f

typedef _Float16 f16;
typedef __attribute__((ext_vector_type(16))) _Float16 v16h;
typedef __attribute__((ext_vector_type(8)))  float    v8f;

union FragU { v16h v; uint32_t u32[8]; };

__device__ __forceinline__ v8f wmma_f16(v16h a, v16h b, v8f c) {
  // D = A(16x32 f16) * B(32x16 f16) + C(16x16 f32)
  return __builtin_amdgcn_wmma_f32_16x16x32_f16(false, a, false, b, (short)0, c,
                                                false, false);
}

// ---------------------------------------------------------------------------
// Fragment loaders following CDNA5 ISA 7.12.2 layouts (wave32).
// A matrix 16x32 (MxK), 16-bit: lane m = lane&15, half = lane>>4;
//   VGPR v: K = (v&3)*2 + (v>=4?16:0) + half*8  (pair K, K+1 packed)
// ---------------------------------------------------------------------------
__device__ __forceinline__ v16h load_frag_a(const f16* base, int ld) {
  const int lane = threadIdx.x & 31;
  const int m = lane & 15, half = lane >> 4;
  FragU f;
#pragma unroll
  for (int vv = 0; vv < 8; ++vv) {
    const int kk = ((vv & 3) * 2) + ((vv & 4) ? 16 : 0) + half * 8;
    f.u32[vv] = *(const uint32_t*)(base + (size_t)m * ld + kk);
  }
  return f.v;
}

// B matrix 32x16 (KxN), 16-bit: lane n = lane&15, half = lane>>4;
//   VGPR v: K = (v&3)*2 + (v>=4?16:0) + half*8. Source row-major [n][k] (K contiguous).
__device__ __forceinline__ v16h load_frag_b(const f16* base, int ld) {
  const int lane = threadIdx.x & 31;
  const int nn = lane & 15, half = lane >> 4;
  FragU f;
#pragma unroll
  for (int vv = 0; vv < 8; ++vv) {
    const int kk = ((vv & 3) * 2) + ((vv & 4) ? 16 : 0) + half * 8;
    f.u32[vv] = *(const uint32_t*)(base + (size_t)nn * ld + kk);
  }
  return f.v;
}

// B loader where per-lane row index (row0 + n) may be out of [0, nrows):
// out-of-range lanes contribute zero columns.
__device__ __forceinline__ v16h load_frag_b_rowclamp(const f16* base, int ld,
                                                     int row0, int nrows) {
  const int lane = threadIdx.x & 31;
  const int nn = lane & 15, half = lane >> 4;
  const int rrow = row0 + nn;
  const bool ok = (rrow >= 0) && (rrow < nrows);
  const f16* p = base + (size_t)(ok ? rrow : 0) * ld;
  FragU f;
#pragma unroll
  for (int vv = 0; vv < 8; ++vv) {
    const uint32_t d = *(const uint32_t*)(p + ((vv & 3) * 2) + ((vv & 4) ? 16 : 0) + half * 8);
    f.u32[vv] = ok ? d : 0u;
  }
  return f.v;
}

// C/D 16x16 f32: lane col n = lane&15, VGPR r -> row m = r + 8*(lane>>4).

// ---------------------------------------------------------------------------
// GEMM core: one wave computes a 32(M) x 64(N) strip of C = A * B^T.
// k-loop unrolled by 2 with ping-pong fragment sets (no register copies, no
// WMMA->VALU WAR hazards) + global_prefetch_b8 two steps ahead.
// A: [M][K] f16 row-major, B: [N][K] f16 row-major. K % 64 == 0, K >= 64.
// ---------------------------------------------------------------------------
__device__ __forceinline__ void gemm_core(const f16* __restrict__ A,
                                          const f16* __restrict__ B,
                                          int K, int lda, int ldb,
                                          int row0, int col0, v8f acc[2][4]) {
  const f16* aP0 = A + (size_t)row0 * lda;
  const f16* aP1 = aP0 + (size_t)16 * lda;
  const f16* bP0 = B + (size_t)col0 * ldb;
  const f16* bP1 = bP0 + (size_t)16 * ldb;
  const f16* bP2 = bP0 + (size_t)32 * ldb;
  const f16* bP3 = bP0 + (size_t)48 * ldb;

  // set X holds fragments for k0, set Y for k0+32
  v16h xa0 = load_frag_a(aP0, lda);
  v16h xa1 = load_frag_a(aP1, lda);
  v16h xb0 = load_frag_b(bP0, ldb);
  v16h xb1 = load_frag_b(bP1, ldb);
  v16h xb2 = load_frag_b(bP2, ldb);
  v16h xb3 = load_frag_b(bP3, ldb);
  v16h ya0, ya1, yb0, yb1, yb2, yb3;

  int k0 = 0;
  for (; k0 + 64 < K; k0 += 64) {
    if (k0 + 128 < K) {
      __builtin_prefetch(aP0 + k0 + 128, 0, 1);
      __builtin_prefetch(aP1 + k0 + 128, 0, 1);
      __builtin_prefetch(bP0 + k0 + 128, 0, 1);
      __builtin_prefetch(bP1 + k0 + 128, 0, 1);
      __builtin_prefetch(bP2 + k0 + 128, 0, 1);
      __builtin_prefetch(bP3 + k0 + 128, 0, 1);
    }
    // load set Y (k0+32), then consume set X
    ya0 = load_frag_a(aP0 + k0 + 32, lda);
    ya1 = load_frag_a(aP1 + k0 + 32, lda);
    yb0 = load_frag_b(bP0 + k0 + 32, ldb);
    yb1 = load_frag_b(bP1 + k0 + 32, ldb);
    yb2 = load_frag_b(bP2 + k0 + 32, ldb);
    yb3 = load_frag_b(bP3 + k0 + 32, ldb);

    acc[0][0] = wmma_f16(xa0, xb0, acc[0][0]);
    acc[1][0] = wmma_f16(xa1, xb0, acc[1][0]);
    acc[0][1] = wmma_f16(xa0, xb1, acc[0][1]);
    acc[1][1] = wmma_f16(xa1, xb1, acc[1][1]);
    acc[0][2] = wmma_f16(xa0, xb2, acc[0][2]);
    acc[1][2] = wmma_f16(xa1, xb2, acc[1][2]);
    acc[0][3] = wmma_f16(xa0, xb3, acc[0][3]);
    acc[1][3] = wmma_f16(xa1, xb3, acc[1][3]);

    // load set X (k0+64), then consume set Y
    xa0 = load_frag_a(aP0 + k0 + 64, lda);
    xa1 = load_frag_a(aP1 + k0 + 64, lda);
    xb0 = load_frag_b(bP0 + k0 + 64, ldb);
    xb1 = load_frag_b(bP1 + k0 + 64, ldb);
    xb2 = load_frag_b(bP2 + k0 + 64, ldb);
    xb3 = load_frag_b(bP3 + k0 + 64, ldb);

    acc[0][0] = wmma_f16(ya0, yb0, acc[0][0]);
    acc[1][0] = wmma_f16(ya1, yb0, acc[1][0]);
    acc[0][1] = wmma_f16(ya0, yb1, acc[0][1]);
    acc[1][1] = wmma_f16(ya1, yb1, acc[1][1]);
    acc[0][2] = wmma_f16(ya0, yb2, acc[0][2]);
    acc[1][2] = wmma_f16(ya1, yb2, acc[1][2]);
    acc[0][3] = wmma_f16(ya0, yb3, acc[0][3]);
    acc[1][3] = wmma_f16(ya1, yb3, acc[1][3]);
  }
  // tail: k0 == K - 64; set X is loaded, load set Y (K - 32)
  ya0 = load_frag_a(aP0 + k0 + 32, lda);
  ya1 = load_frag_a(aP1 + k0 + 32, lda);
  yb0 = load_frag_b(bP0 + k0 + 32, ldb);
  yb1 = load_frag_b(bP1 + k0 + 32, ldb);
  yb2 = load_frag_b(bP2 + k0 + 32, ldb);
  yb3 = load_frag_b(bP3 + k0 + 32, ldb);

  acc[0][0] = wmma_f16(xa0, xb0, acc[0][0]);
  acc[1][0] = wmma_f16(xa1, xb0, acc[1][0]);
  acc[0][1] = wmma_f16(xa0, xb1, acc[0][1]);
  acc[1][1] = wmma_f16(xa1, xb1, acc[1][1]);
  acc[0][2] = wmma_f16(xa0, xb2, acc[0][2]);
  acc[1][2] = wmma_f16(xa1, xb2, acc[1][2]);
  acc[0][3] = wmma_f16(xa0, xb3, acc[0][3]);
  acc[1][3] = wmma_f16(xa1, xb3, acc[1][3]);

  acc[0][0] = wmma_f16(ya0, yb0, acc[0][0]);
  acc[1][0] = wmma_f16(ya1, yb0, acc[1][0]);
  acc[0][1] = wmma_f16(ya0, yb1, acc[0][1]);
  acc[1][1] = wmma_f16(ya1, yb1, acc[1][1]);
  acc[0][2] = wmma_f16(ya0, yb2, acc[0][2]);
  acc[1][2] = wmma_f16(ya1, yb2, acc[1][2]);
  acc[0][3] = wmma_f16(ya0, yb3, acc[0][3]);
  acc[1][3] = wmma_f16(ya1, yb3, acc[1][3]);
}

__device__ __forceinline__ void zero_acc(v8f acc[2][4]) {
#pragma unroll
  for (int mr = 0; mr < 2; ++mr)
#pragma unroll
    for (int c = 0; c < 4; ++c)
#pragma unroll
      for (int r = 0; r < 8; ++r) acc[mr][c][r] = 0.0f;
}

// ---------------------------------------------------------------------------
// Conversion kernels
// ---------------------------------------------------------------------------
__global__ void cvt_kernel(const float* __restrict__ in, f16* __restrict__ out,
                           size_t n) {
  size_t i = (size_t)blockIdx.x * blockDim.x + threadIdx.x;
  if (i < n) out[i] = (f16)in[i];
}

__global__ void build_ctx_kernel(const float* __restrict__ mem,
                                 const float* __restrict__ x,
                                 f16* __restrict__ ctx) {
  size_t i = (size_t)blockIdx.x * blockDim.x + threadIdx.x;
  const size_t memN = (size_t)MLEN * BSZ * D_MODEL;
  const size_t totN = (size_t)RLEN * BSZ * D_MODEL;
  if (i < totN) ctx[i] = (f16)((i < memN) ? mem[i] : x[i - memN]);
}

// ---------------------------------------------------------------------------
// QKV projection: C[4096 x 3072] = ctx * Wqkv^T, split epilogue.
//   cols [0,1024):    q rows (t>=MLEN) -> qu = q+u, qv = q+v   (f16)
//   cols [1024,2048): wk  (klen,bsz,1024) f16
//   cols [2048,3072): wv stored transposed per (b,h): [b][h][d][t] f16
// grid: (M/128, N/64), block: 128 (4 waves stacked in M, 32 rows each)
// ---------------------------------------------------------------------------
__global__ __launch_bounds__(128)
void gemm_qkv_kernel(const f16* __restrict__ A, const f16* __restrict__ B,
                     const float* __restrict__ uvec, const float* __restrict__ vvec,
                     f16* __restrict__ qu, f16* __restrict__ qv,
                     f16* __restrict__ wk, f16* __restrict__ wvT) {
  const int wave = threadIdx.x >> 5;
  const int lane = threadIdx.x & 31;
  const int nn = lane & 15, half = lane >> 4;
  const int row0 = blockIdx.x * 128 + wave * 32;
  const int col0 = blockIdx.y * 64;

  v8f acc[2][4];
  zero_acc(acc);
  gemm_core(A, B, D_MODEL, D_MODEL, D_MODEL, row0, col0, acc);

#pragma unroll
  for (int mr = 0; mr < 2; ++mr) {
#pragma unroll
    for (int c = 0; c < 4; ++c) {
      const int gcol = col0 + c * 16 + nn;
#pragma unroll
      for (int r = 0; r < 8; ++r) {
        const int grow = row0 + mr * 16 + r + 8 * half;  // = t*BSZ + bb
        const int t = grow >> 1;                         // BSZ == 2
        const int bb = grow & 1;
        const float val = acc[mr][c][r];
        if (gcol < D_MODEL) {
          if (t >= MLEN) {
            const size_t o = ((size_t)(t - MLEN) * BSZ + bb) * D_MODEL + gcol;
            qu[o] = (f16)(val + uvec[gcol]);
            qv[o] = (f16)(val + vvec[gcol]);
          }
        } else if (gcol < 2 * D_MODEL) {
          wk[(size_t)grow * D_MODEL + (gcol - D_MODEL)] = (f16)val;
        } else {
          const int c2 = gcol - 2 * D_MODEL;
          const int h = c2 >> 6, dd = c2 & 63;
          wvT[(((size_t)bb * N_HEADS + h) * HEAD_DIM + dd) * RLEN + t] = (f16)val;
        }
      }
    }
  }
}

// Plain GEMM, f16 output: C[M x N] = A * B^T. grid (M/128, N/64), block 128.
__global__ __launch_bounds__(128)
void gemm_f16_kernel(const f16* __restrict__ A, const f16* __restrict__ B,
                     f16* __restrict__ C, int K, int lda, int ldb, int N) {
  const int wave = threadIdx.x >> 5;
  const int lane = threadIdx.x & 31;
  const int nn = lane & 15, half = lane >> 4;
  const int row0 = blockIdx.x * 128 + wave * 32;
  const int col0 = blockIdx.y * 64;
  v8f acc[2][4];
  zero_acc(acc);
  gemm_core(A, B, K, lda, ldb, row0, col0, acc);
#pragma unroll
  for (int mr = 0; mr < 2; ++mr)
#pragma unroll
    for (int c = 0; c < 4; ++c)
#pragma unroll
      for (int r = 0; r < 8; ++r)
        C[(size_t)(row0 + mr * 16 + r + 8 * half) * N + (col0 + c * 16 + nn)] =
            (f16)acc[mr][c][r];
}

// Plain GEMM, f32 output
__global__ __launch_bounds__(128)
void gemm_f32_kernel(const f16* __restrict__ A, const f16* __restrict__ B,
                     float* __restrict__ C, int K, int lda, int ldb, int N) {
  const int wave = threadIdx.x >> 5;
  const int lane = threadIdx.x & 31;
  const int nn = lane & 15, half = lane >> 4;
  const int row0 = blockIdx.x * 128 + wave * 32;
  const int col0 = blockIdx.y * 64;
  v8f acc[2][4];
  zero_acc(acc);
  gemm_core(A, B, K, lda, ldb, row0, col0, acc);
#pragma unroll
  for (int mr = 0; mr < 2; ++mr)
#pragma unroll
    for (int c = 0; c < 4; ++c)
#pragma unroll
      for (int r = 0; r < 8; ++r)
        C[(size_t)(row0 + mr * 16 + r + 8 * half) * N + (col0 + c * 16 + nn)] =
            acc[mr][c][r];
}

// ---------------------------------------------------------------------------
// Fused relative attention (flash-style online softmax).
// grid = (QLEN/16, N_HEADS, BSZ), block = 32 (one wave).
// rel_shift: out[a,c] = BD_raw[a, c-a+QLEN-1]  (OOB raw col -> 0; the
// softmax-masked tail region is zeroed).
// ---------------------------------------------------------------------------
__global__ __launch_bounds__(32)
void attn_kernel(const f16* __restrict__ qu, const f16* __restrict__ qv,
                 const f16* __restrict__ wk, const f16* __restrict__ rk,
                 const f16* __restrict__ wvT, f16* __restrict__ avec) {
  const int a0 = blockIdx.x * 16;
  const int h  = blockIdx.y;
  const int b  = blockIdx.z;
  const int lane = threadIdx.x & 31;
  const int nn = lane & 15, half = lane >> 4;
  const float scale = 0.125f;           // 1/sqrt(64)
  const int klen = RLEN;
  const int ldq = BSZ * D_MODEL;

  __shared__ float bdraw[16][33];       // 16 rows x 32-band (+pad)
  __shared__ f16   pbuf[16][32];

  const f16* quBase = qu + ((size_t)a0 * BSZ + b) * D_MODEL + h * HEAD_DIM;
  const f16* qvBase = qv + ((size_t)a0 * BSZ + b) * D_MODEL + h * HEAD_DIM;
  const f16* kBase  = wk + (size_t)b * D_MODEL + h * HEAD_DIM;   // + j*ldq
  const f16* rBase  = rk + h * HEAD_DIM;                          // + j*D_MODEL
  const f16* vBase  = wvT + ((size_t)b * N_HEADS + h) * HEAD_DIM * (size_t)RLEN;

  const v16h qu0 = load_frag_a(quBase, ldq);
  const v16h qu1 = load_frag_a(quBase + 32, ldq);
  const v16h qv0 = load_frag_a(qvBase, ldq);
  const v16h qv1 = load_frag_a(qvBase + 32, ldq);

  v8f o0, o1, o2, o3;
  float mrun[8], lrun[8], sreg[2][8];
#pragma unroll
  for (int r = 0; r < 8; ++r) {
    o0[r] = 0.0f; o1[r] = 0.0f; o2[r] = 0.0f; o3[r] = 0.0f;
    mrun[r] = -3.0e38f; lrun[r] = 0.0f;
  }

  for (int j0 = 0; j0 < klen; j0 += 32) {
    // prefetch next j-tile's K rows and V columns (global_prefetch_b8)
    if (j0 + 32 < klen) {
      __builtin_prefetch(kBase + (size_t)(j0 + 32 + lane) * ldq, 0, 1);
      __builtin_prefetch(vBase + (size_t)(lane & 3) * 16 * RLEN +
                             (size_t)(lane >> 2) * 8 * RLEN + j0 + 32, 0, 1);
      const int pr = j0 + 32 + QLEN - a0 - 16 + lane;
      if (pr >= 0 && pr < klen)
        __builtin_prefetch(rBase + (size_t)pr * D_MODEL, 0, 1);
    }
    // ------- scores for two 16-column subtiles -------
    for (int sub = 0; sub < 2; ++sub) {
      const int j0s = j0 + sub * 16;
      // AC = (q+u) . K^T
      v8f ac;
#pragma unroll
      for (int r = 0; r < 8; ++r) ac[r] = 0.0f;
      {
        const v16h bk0 = load_frag_b(kBase + (size_t)j0s * ldq, ldq);
        const v16h bk1 = load_frag_b(kBase + (size_t)j0s * ldq + 32, ldq);
        ac = wmma_f16(qu0, bk0, ac);
        ac = wmma_f16(qu1, bk1, ac);
      }
      // BD raw band: raw cols [band, band+31], band = j0s + QLEN - a0 - 16
      const int band = j0s + QLEN - a0 - 16;
#pragma unroll
      for (int t = 0; t < 2; ++t) {
        const int raw0 = band + t * 16;
        v8f bd;
#pragma unroll
        for (int r = 0; r < 8; ++r) bd[r] = 0.0f;
        const v16h br0 = load_frag_b_rowclamp(rBase,      D_MODEL, raw0, klen);
        const v16h br1 = load_frag_b_rowclamp(rBase + 32, D_MODEL, raw0, klen);
        bd = wmma_f16(qv0, br0, bd);
        bd = wmma_f16(qv1, br1, bd);
#pragma unroll
        for (int r = 0; r < 8; ++r)
          bdraw[r + 8 * half][t * 16 + nn] = bd[r];
      }
      __syncthreads();
#pragma unroll
      for (int r = 0; r < 8; ++r) {
        const int m = r + 8 * half;
        const float sbd = bdraw[m][15 + nn - m];   // diagonal rel-shift gather
        sreg[sub][r] = (ac[r] + sbd) * scale;
      }
      __syncthreads();
    }

    // ------- online softmax over these 32 columns -------
    float rmax[8], rsum[8], alpha[8];
#pragma unroll
    for (int r = 0; r < 8; ++r) rmax[r] = fmaxf(sreg[0][r], sreg[1][r]);
#pragma unroll
    for (int off = 1; off < 16; off <<= 1)
#pragma unroll
      for (int r = 0; r < 8; ++r)
        rmax[r] = fmaxf(rmax[r], __shfl_xor(rmax[r], off, 32));
#pragma unroll
    for (int r = 0; r < 8; ++r) {
      const float nm = fmaxf(mrun[r], rmax[r]);
      alpha[r] = __expf(mrun[r] - nm);
      mrun[r] = nm;
      const float p0 = __expf(sreg[0][r] - nm);
      const float p1 = __expf(sreg[1][r] - nm);
      sreg[0][r] = p0; sreg[1][r] = p1;
      rsum[r] = p0 + p1;
    }
#pragma unroll
    for (int off = 1; off < 16; off <<= 1)
#pragma unroll
      for (int r = 0; r < 8; ++r)
        rsum[r] += __shfl_xor(rsum[r], off, 32);
#pragma unroll
    for (int r = 0; r < 8; ++r) {
      lrun[r] = lrun[r] * alpha[r] + rsum[r];
      o0[r] *= alpha[r]; o1[r] *= alpha[r];
      o2[r] *= alpha[r]; o3[r] *= alpha[r];
    }

    // ------- O += P * V  (stage P -> LDS, reload as A fragment) -------
#pragma unroll
    for (int sub = 0; sub < 2; ++sub)
#pragma unroll
      for (int r = 0; r < 8; ++r)
        pbuf[r + 8 * half][sub * 16 + nn] = (f16)sreg[sub][r];
    __syncthreads();
    FragU pf;
#pragma unroll
    for (int vv = 0; vv < 8; ++vv) {
      const int kk = ((vv & 3) * 2) + ((vv & 4) ? 16 : 0) + half * 8;
      pf.u32[vv] = *(const uint32_t*)&pbuf[nn][kk];
    }
    const v16h pfrag = pf.v;
    {
      const v16h bv0 = load_frag_b(vBase + (size_t)0  * RLEN + j0, RLEN);
      const v16h bv1 = load_frag_b(vBase + (size_t)16 * RLEN + j0, RLEN);
      const v16h bv2 = load_frag_b(vBase + (size_t)32 * RLEN + j0, RLEN);
      const v16h bv3 = load_frag_b(vBase + (size_t)48 * RLEN + j0, RLEN);
      o0 = wmma_f16(pfrag, bv0, o0);
      o1 = wmma_f16(pfrag, bv1, o1);
      o2 = wmma_f16(pfrag, bv2, o2);
      o3 = wmma_f16(pfrag, bv3, o3);
    }
    __syncthreads();
  }

  // normalize and store attn_vec (qlen, bsz, 1024) f16
#pragma unroll
  for (int r = 0; r < 8; ++r) {
    const float inv = 1.0f / lrun[r];
    const size_t rowOff = ((size_t)(a0 + r + 8 * half) * BSZ + b) * D_MODEL +
                          h * HEAD_DIM + nn;
    avec[rowOff +  0] = (f16)(o0[r] * inv);
    avec[rowOff + 16] = (f16)(o1[r] * inv);
    avec[rowOff + 32] = (f16)(o2[r] * inv);
    avec[rowOff + 48] = (f16)(o3[r] * inv);
  }
}

// ---------------------------------------------------------------------------
// Residual + LayerNorm. One block (256 threads) per row of 1024.
// ---------------------------------------------------------------------------
__global__ __launch_bounds__(256)
void add_ln_kernel(const float* __restrict__ x, const float* __restrict__ ao,
                   const float* __restrict__ gamma, const float* __restrict__ beta,
                   float* __restrict__ out) {
  const int row = blockIdx.x;
  const int tid = threadIdx.x;
  const float* xr = x + (size_t)row * D_MODEL;
  const float* ar = ao + (size_t)row * D_MODEL;
  __shared__ float rs[256], rs2[256];
  float yv[4], s = 0.0f, s2 = 0.0f;
#pragma unroll
  for (int e = 0; e < 4; ++e) {
    const int i = tid + e * 256;
    const float y = xr[i] + ar[i];
    yv[e] = y; s += y; s2 += y * y;
  }
  rs[tid] = s; rs2[tid] = s2;
  __syncthreads();
  for (int o = 128; o > 0; o >>= 1) {
    if (tid < o) { rs[tid] += rs[tid + o]; rs2[tid] += rs2[tid + o]; }
    __syncthreads();
  }
  const float mean = rs[0] * (1.0f / D_MODEL);
  const float var  = rs2[0] * (1.0f / D_MODEL) - mean * mean;
  const float inv  = rsqrtf(var + LN_EPS);
#pragma unroll
  for (int e = 0; e < 4; ++e) {
    const int i = tid + e * 256;
    out[(size_t)row * D_MODEL + i] = (yv[e] - mean) * inv * gamma[i] + beta[i];
  }
}

// ---------------------------------------------------------------------------
extern "C" void kernel_launch(void* const* d_in, const int* in_sizes, int n_in,
                              void* d_out, int out_size, void* d_ws, size_t ws_size,
                              hipStream_t stream) {
  (void)in_sizes; (void)n_in; (void)out_size; (void)ws_size;
  const float* x     = (const float*)d_in[0];
  const float* r     = (const float*)d_in[1];
  const float* u     = (const float*)d_in[2];
  const float* v     = (const float*)d_in[3];
  const float* mem   = (const float*)d_in[4];
  const float* Wqkv  = (const float*)d_in[5];
  const float* Wr    = (const float*)d_in[6];
  const float* Wout  = (const float*)d_in[7];
  const float* gamma = (const float*)d_in[8];
  const float* beta  = (const float*)d_in[9];
  float* out = (float*)d_out;

  char* ws = (char*)d_ws;
  size_t off = 0;
  auto carve = [&](size_t bytes) -> void* {
    void* p = ws + off;
    off = (off + bytes + 255) & ~(size_t)255;
    return p;
  };
  f16* ctx16  = (f16*)carve((size_t)RLEN * BSZ * D_MODEL * sizeof(f16));
  f16* wqkv16 = (f16*)carve((size_t)3 * D_MODEL * D_MODEL * sizeof(f16));
  f16* wr16   = (f16*)carve((size_t)D_MODEL * D_MODEL * sizeof(f16));
  f16* wout16 = (f16*)carve((size_t)D_MODEL * D_MODEL * sizeof(f16));
  f16* r16    = (f16*)carve((size_t)RLEN * D_MODEL * sizeof(f16));
  f16* qu16   = (f16*)carve((size_t)QLEN * BSZ * D_MODEL * sizeof(f16));
  f16* qv16   = (f16*)carve((size_t)QLEN * BSZ * D_MODEL * sizeof(f16));
  f16* wk16   = (f16*)carve((size_t)RLEN * BSZ * D_MODEL * sizeof(f16));
  f16* wvT16  = (f16*)carve((size_t)RLEN * BSZ * D_MODEL * sizeof(f16));
  f16* rk16   = (f16*)carve((size_t)RLEN * D_MODEL * sizeof(f16));
  f16* avec16 = (f16*)carve((size_t)QLEN * BSZ * D_MODEL * sizeof(f16));
  float* aout = (float*)carve((size_t)QLEN * BSZ * D_MODEL * sizeof(float));

  // --- stage 0: f32 -> f16 conversions ---
  {
    const size_t nctx = (size_t)RLEN * BSZ * D_MODEL;
    build_ctx_kernel<<<(unsigned)((nctx + 255) / 256), 256, 0, stream>>>(mem, x, ctx16);
    const size_t nqkv = (size_t)3 * D_MODEL * D_MODEL;
    cvt_kernel<<<(unsigned)((nqkv + 255) / 256), 256, 0, stream>>>(Wqkv, wqkv16, nqkv);
    const size_t nsq = (size_t)D_MODEL * D_MODEL;
    cvt_kernel<<<(unsigned)((nsq + 255) / 256), 256, 0, stream>>>(Wr, wr16, nsq);
    cvt_kernel<<<(unsigned)((nsq + 255) / 256), 256, 0, stream>>>(Wout, wout16, nsq);
    const size_t nr = (size_t)RLEN * D_MODEL;
    cvt_kernel<<<(unsigned)((nr + 255) / 256), 256, 0, stream>>>(r, r16, nr);
  }

  // --- stage 1: QKV projection (M=4096, N=3072, K=1024) ---
  gemm_qkv_kernel<<<dim3((RLEN * BSZ) / 128, (3 * D_MODEL) / 64), 128, 0, stream>>>(
      ctx16, wqkv16, u, v, qu16, qv16, wk16, wvT16);

  // --- stage 2: rk = r @ W_r^T (M=2048, N=1024, K=1024) ---
  gemm_f16_kernel<<<dim3(RLEN / 128, D_MODEL / 64), 128, 0, stream>>>(
      r16, wr16, rk16, D_MODEL, D_MODEL, D_MODEL, D_MODEL);

  // --- stage 3: fused relative attention ---
  attn_kernel<<<dim3(QLEN / 16, N_HEADS, BSZ), 32, 0, stream>>>(
      qu16, qv16, wk16, rk16, wvT16, avec16);

  // --- stage 4: output projection (M=2048, N=1024, K=1024) ---
  gemm_f32_kernel<<<dim3((QLEN * BSZ) / 128, D_MODEL / 64), 128, 0, stream>>>(
      avec16, wout16, aout, D_MODEL, D_MODEL, D_MODEL, D_MODEL);

  // --- stage 5: residual + LayerNorm ---
  add_ln_kernel<<<QLEN * BSZ, 256, 0, stream>>>(x, aout, gamma, beta, out);
}
